// EfficientAttention_81896436400171
// MI455X (gfx1250) — matrix operand order repaired and checked by reading, not compiled, not run on hardware
//
#include <hip/hip_runtime.h>
#include <hip/hip_bf16.h>

typedef __bf16 bf16_t;
typedef __attribute__((ext_vector_type(16))) __bf16 v16bf;
typedef __attribute__((ext_vector_type(8)))  __bf16 v8bf;
typedef __attribute__((ext_vector_type(8)))  float  v8f;
typedef __attribute__((ext_vector_type(4)))  int    v4i;

typedef __attribute__((address_space(1))) v4i gv4i;   // global int4
typedef __attribute__((address_space(3))) v4i lv4i;   // LDS int4

#define DIMC     1024
#define HEADS    16
#define HEAD_DIM 64
#define WINDOW   128
#define SEQ      2048
#define BATCH    2

#define WMMA_BF16(A, B, C) \
  __builtin_amdgcn_wmma_f32_16x16x32_bf16(false, (A), false, (B), (short)0, (C), false, false)

// ---------------- async global -> LDS (CDNA5 GLOBAL_LOAD_ASYNC_TO_LDS_B128) --
#if __has_builtin(__builtin_amdgcn_global_load_async_to_lds_b128)
#define HAS_ASYNC_LDS 1
#endif

__device__ __forceinline__ void async_cp16(const bf16_t* g, bf16_t* l) {
#ifdef HAS_ASYNC_LDS
  __builtin_amdgcn_global_load_async_to_lds_b128((gv4i*)g, (lv4i*)l, 0, 0);
#else
  *(uint4*)l = *(const uint4*)g;
#endif
}

__device__ __forceinline__ void wait_async0() {
#if __has_builtin(__builtin_amdgcn_s_wait_asynccnt)
  __builtin_amdgcn_s_wait_asynccnt(0);
#else
  asm volatile("s_wait_asynccnt 0x0" ::: "memory");
#endif
}

// ---- fragment loaders (ISA 7.12.2 layouts, wave32) ----
__device__ __forceinline__ v16bf load_A_frag(const bf16_t* base, int stride) {
  const int lane = threadIdx.x & 31;
  const int row  = lane & 15;
  const int hf   = lane >> 4;
  const bf16_t* p = base + row * stride + hf * 8;
  v8bf lo = *(const v8bf*)(p);
  v8bf hi = *(const v8bf*)(p + 16);
  return __builtin_shufflevector(lo, hi, 0,1,2,3,4,5,6,7,8,9,10,11,12,13,14,15);
}

__device__ __forceinline__ v16bf load_B_frag(const bf16_t* base, int stride) {
  const int lane = threadIdx.x & 31;
  const int row  = lane & 15;
  const int hf   = lane >> 4;
  const bf16_t* p = base + row * stride + hf * 8;
  v8bf lo = *(const v8bf*)(p);
  v8bf hi = *(const v8bf*)(p + 16);
  return __builtin_shufflevector(lo, hi, 0,1,2,3,4,5,6,7,8,9,10,11,12,13,14,15);
}

__device__ __forceinline__ unsigned pack2_bf16(float a, float b) {
  union { bf16_t h[2]; unsigned u; } v;
  v.h[0] = (bf16_t)a; v.h[1] = (bf16_t)b;
  return v.u;
}

// =====================================================================
// fp32 -> bf16 conversion pass (bandwidth bound)
// n must be a multiple of 2048; 8 elements/thread.
// =====================================================================
__global__ __launch_bounds__(256) void cvt_bf16(const float* __restrict__ in,
                                                bf16_t* __restrict__ out, int n) {
  const int i = (blockIdx.x * 256 + threadIdx.x) * 8;
  if (i >= n) return;
  float4 a = *(const float4*)(in + i);
  float4 b = *(const float4*)(in + i + 4);
  uint4 u;
  u.x = pack2_bf16(a.x, a.y);
  u.y = pack2_bf16(a.z, a.w);
  u.z = pack2_bf16(b.x, b.y);
  u.w = pack2_bf16(b.z, b.w);
  *(uint4*)(out + i) = u;
}

// =====================================================================
// GEMM: C[M,N] = A[M,K] * W[N,K]^T, bf16 inputs, f32 accumulate.
// 128x128 tile, BK=32, 8 waves, double-buffered LDS filled by async copies.
// MODE 0: epilogue scatters bf16 Q[B,H,T,hd], K[B,H,T,hd], V^T[B,H,hd,T]
// MODE 1: epilogue writes fp32 Out + bias
// =====================================================================
template <int MODE>
__global__ __launch_bounds__(256) void gemm_wmma(
    const bf16_t* __restrict__ A, const bf16_t* __restrict__ W,
    const float* __restrict__ bias,
    bf16_t* __restrict__ Qo, bf16_t* __restrict__ Ko, bf16_t* __restrict__ VTo,
    float* __restrict__ Out, int M, int N, int K)
{
  __shared__ bf16_t sA[2][128 * 40];
  __shared__ bf16_t sB[2][128 * 40];

  const int tid = threadIdx.x;
  const int wid = tid >> 5;
  const int wm  = wid & 3;   // 4 wave-rows -> 32 rows each
  const int wn  = wid >> 2;  // 2 wave-cols -> 64 cols each
  const int m0  = blockIdx.y * 128;
  const int n0  = blockIdx.x * 128;

  v8f acc[2][4];
  #pragma unroll
  for (int i = 0; i < 2; ++i)
    #pragma unroll
    for (int j = 0; j < 4; ++j) acc[i][j] = (v8f){0.f,0.f,0.f,0.f,0.f,0.f,0.f,0.f};

  const int srow = tid >> 1;         // 0..127
  const int scol = (tid & 1) * 16;   // 0 or 16

  auto stage = [&](int k0, int b) {
    const bf16_t* ga = A + (size_t)(m0 + srow) * K + k0 + scol;
    const bf16_t* gw = W + (size_t)(n0 + srow) * K + k0 + scol;
    bf16_t* la = &sA[b][srow * 40 + scol];
    bf16_t* lw = &sB[b][srow * 40 + scol];
    async_cp16(ga,     la);
    async_cp16(ga + 8, la + 8);
    async_cp16(gw,     lw);
    async_cp16(gw + 8, lw + 8);
    if (k0 + 64 < K) {
      __builtin_prefetch(ga + 64, 0, 1);
      __builtin_prefetch(gw + 64, 0, 1);
    }
  };

  stage(0, 0);
  int buf = 0;
  for (int k0 = 0; k0 < K; k0 += 32) {
    wait_async0();
    __syncthreads();
    if (k0 + 32 < K) stage(k0 + 32, buf ^ 1);

    v16bf a0 = load_A_frag(&sA[buf][(wm * 32)      * 40], 40);
    v16bf a1 = load_A_frag(&sA[buf][(wm * 32 + 16) * 40], 40);
    #pragma unroll
    for (int ni = 0; ni < 4; ++ni) {
      v16bf bfr = load_B_frag(&sB[buf][(wn * 64 + ni * 16) * 40], 40);
      acc[0][ni] = WMMA_BF16(a0, bfr, acc[0][ni]);
      acc[1][ni] = WMMA_BF16(a1, bfr, acc[1][ni]);
    }
    buf ^= 1;
  }

  // ---- epilogue (C layout: VGPR r, lane L: m = r + 8*(L>>4), n = L&15) ----
  const int lane = tid & 31;
  const int hf   = lane >> 4;
  const int col  = lane & 15;
  #pragma unroll
  for (int mi = 0; mi < 2; ++mi)
    #pragma unroll
    for (int ni = 0; ni < 4; ++ni)
      #pragma unroll
      for (int r = 0; r < 8; ++r) {
        const int m = m0 + wm * 32 + mi * 16 + r + hf * 8;
        const int n = n0 + wn * 64 + ni * 16 + col;
        const float v = acc[mi][ni][r];
        if (MODE == 1) {
          Out[(size_t)m * N + n] = v + bias[n];
        } else {
          const int bb = m >> 11;          // /SEQ
          const int t  = m & (SEQ - 1);
          const bf16_t bv = (bf16_t)v;
          if (n < DIMC) {
            const int h = n >> 6, d = n & 63;
            Qo[(((size_t)bb * HEADS + h) * SEQ + t) * HEAD_DIM + d] = bv;
          } else if (n < 2 * DIMC) {
            const int nn = n - DIMC, h = nn >> 6, d = nn & 63;
            Ko[(((size_t)bb * HEADS + h) * SEQ + t) * HEAD_DIM + d] = bv;
          } else {
            const int nn = n - 2 * DIMC, h = nn >> 6, d = nn & 63;
            VTo[(((size_t)bb * HEADS + h) * HEAD_DIM + d) * SEQ + t] = bv;
          }
        }
      }
}

// =====================================================================
// Sliding-window attention, flash-style online softmax.
// 128 threads = 4 waves; each wave owns a 16-row query tile.
// =====================================================================
__global__ __launch_bounds__(128) void attn_wmma(
    const bf16_t* __restrict__ Q, const bf16_t* __restrict__ K,
    const bf16_t* __restrict__ VT, bf16_t* __restrict__ AO)
{
  __shared__ bf16_t sP[4 * 16 * 40];   // per-wave P staging (C->A layout)

  const int tid  = threadIdx.x;
  const int wid  = tid >> 5;
  const int lane = tid & 31;
  const int hf   = lane >> 4;
  const int col  = lane & 15;

  const int qblk = blockIdx.x % (SEQ / 64);
  const int bh   = blockIdx.x / (SEQ / 64);   // b*HEADS + h
  const int b    = bh / HEADS;
  const int h    = bh % HEADS;
  const int q0   = qblk * 64 + wid * 16;

  const bf16_t* Qh = Q  + (size_t)bh * SEQ * HEAD_DIM;
  const bf16_t* Kh = K  + (size_t)bh * SEQ * HEAD_DIM;
  const bf16_t* Vh = VT + (size_t)bh * HEAD_DIM * SEQ;

  const float scale = 0.125f;   // 1/sqrt(64)

  const v16bf qa0 = load_A_frag(Qh + (size_t)q0 * HEAD_DIM,      HEAD_DIM);
  const v16bf qa1 = load_A_frag(Qh + (size_t)q0 * HEAD_DIM + 32, HEAD_DIM);

  float mrow[8], lrow[8];
  v8f o[4];
  #pragma unroll
  for (int r = 0; r < 8; ++r) { mrow[r] = -1e30f; lrow[r] = 0.f; }
  #pragma unroll
  for (int c4 = 0; c4 < 4; ++c4) o[c4] = (v8f){0.f,0.f,0.f,0.f,0.f,0.f,0.f,0.f};

  int kstart = q0 - WINDOW; if (kstart < 0) kstart = 0; kstart &= ~31;
  int kend   = q0 + 16 + WINDOW; if (kend > SEQ) kend = SEQ;

  bf16_t* pw = &sP[wid * 16 * 40];

  for (int ks = kstart; ks < kend; ks += 32) {
    const bf16_t* Kp = Kh + (size_t)ks * HEAD_DIM;
    const v16bf kb00 = load_B_frag(Kp,                      HEAD_DIM);
    const v16bf kb01 = load_B_frag(Kp + 32,                 HEAD_DIM);
    const v16bf kb10 = load_B_frag(Kp + 16 * HEAD_DIM,      HEAD_DIM);
    const v16bf kb11 = load_B_frag(Kp + 16 * HEAD_DIM + 32, HEAD_DIM);

    v8f s0 = (v8f){0.f,0.f,0.f,0.f,0.f,0.f,0.f,0.f};
    v8f s1 = s0;
    s0 = WMMA_BF16(qa0, kb00, s0);
    s0 = WMMA_BF16(qa1, kb01, s0);
    s1 = WMMA_BF16(qa0, kb10, s1);
    s1 = WMMA_BF16(qa1, kb11, s1);

    float corr[8];
    #pragma unroll
    for (int r = 0; r < 8; ++r) {
      const int qrow = q0 + r + hf * 8;
      const int kA = ks + col;
      const int kB = ks + 16 + col;
      float v0 = s0[r] * scale;
      float v1 = s1[r] * scale;
      if (kA < qrow - WINDOW || kA > qrow + WINDOW) v0 = -1e30f;
      if (kB < qrow - WINDOW || kB > qrow + WINDOW) v1 = -1e30f;
      float mr = fmaxf(v0, v1);
      mr = fmaxf(mr, __shfl_xor(mr, 1, 32));
      mr = fmaxf(mr, __shfl_xor(mr, 2, 32));
      mr = fmaxf(mr, __shfl_xor(mr, 4, 32));
      mr = fmaxf(mr, __shfl_xor(mr, 8, 32));
      const float mnew = fmaxf(mrow[r], mr);
      const float c    = __expf(mrow[r] - mnew);
      const float p0   = __expf(v0 - mnew);
      const float p1   = __expf(v1 - mnew);
      float ps = p0 + p1;
      ps += __shfl_xor(ps, 1, 32);
      ps += __shfl_xor(ps, 2, 32);
      ps += __shfl_xor(ps, 4, 32);
      ps += __shfl_xor(ps, 8, 32);
      lrow[r] = lrow[r] * c + ps;
      mrow[r] = mnew;
      corr[r] = c;
      s0[r] = p0; s1[r] = p1;
    }
    #pragma unroll
    for (int c4 = 0; c4 < 4; ++c4)
      #pragma unroll
      for (int r = 0; r < 8; ++r) o[c4][r] *= corr[r];

    // P (C layout) -> LDS -> A fragment
    #pragma unroll
    for (int r = 0; r < 8; ++r) {
      pw[(r + hf * 8) * 40 + col]      = (bf16_t)s0[r];
      pw[(r + hf * 8) * 40 + 16 + col] = (bf16_t)s1[r];
    }
    const v16bf pa = load_A_frag(pw, 40);

    #pragma unroll
    for (int c4 = 0; c4 < 4; ++c4) {
      const v16bf vb = load_B_frag(Vh + (size_t)(c4 * 16) * SEQ + ks, SEQ);
      o[c4] = WMMA_BF16(pa, vb, o[c4]);
    }
  }

  float inv[8];
  #pragma unroll
  for (int r = 0; r < 8; ++r) inv[r] = 1.0f / lrow[r];

  bf16_t* aop = AO + (size_t)b * SEQ * DIMC + (size_t)h * HEAD_DIM;
  #pragma unroll
  for (int c4 = 0; c4 < 4; ++c4)
    #pragma unroll
    for (int r = 0; r < 8; ++r) {
      const int qrow = q0 + r + hf * 8;
      aop[(size_t)qrow * DIMC + c4 * 16 + col] = (bf16_t)(o[c4][r] * inv[r]);
    }
}

// =====================================================================
extern "C" void kernel_launch(void* const* d_in, const int* in_sizes, int n_in,
                              void* d_out, int out_size, void* d_ws, size_t ws_size,
                              hipStream_t stream) {
  const float* x     = (const float*)d_in[0];
  const float* Wqkv  = (const float*)d_in[1];
  const float* Wproj = (const float*)d_in[2];
  const float* bproj = (const float*)d_in[3];
  float* out = (float*)d_out;

  const size_t elems = (size_t)BATCH * SEQ * DIMC;       // 4 Mi
  const int nX  = BATCH * SEQ * DIMC;                    // 4,194,304
  const int nWq = 3 * DIMC * DIMC;                       // 3,145,728
  const int nWp = DIMC * DIMC;                           // 1,048,576

  bf16_t* qb  = (bf16_t*)d_ws;
  bf16_t* kb  = qb  + elems;
  bf16_t* vtb = kb  + elems;
  bf16_t* ao  = vtb + elems;
  bf16_t* xb  = ao  + elems;
  bf16_t* wqb = xb  + (size_t)nX;
  bf16_t* wpb = wqb + (size_t)nWq;

  const int M = BATCH * SEQ;

  // 0) fp32 -> bf16 conversions
  cvt_bf16<<<dim3(nX  / 2048), 256, 0, stream>>>(x,     xb,  nX);
  cvt_bf16<<<dim3(nWq / 2048), 256, 0, stream>>>(Wqkv,  wqb, nWq);
  cvt_bf16<<<dim3(nWp / 2048), 256, 0, stream>>>(Wproj, wpb, nWp);

  // 1) QKV projection: [4096,1024] x [3072,1024]^T
  gemm_wmma<0><<<dim3((3 * DIMC) / 128, M / 128), 256, 0, stream>>>(
      xb, wqb, nullptr, qb, kb, vtb, nullptr, M, 3 * DIMC, DIMC);

  // 2) sliding-window attention
  attn_wmma<<<dim3(BATCH * HEADS * (SEQ / 64)), 128, 0, stream>>>(qb, kb, vtb, ao);

  // 3) output projection + bias: [4096,1024] x [1024,1024]^T
  gemm_wmma<1><<<dim3(DIMC / 128, M / 128), 256, 0, stream>>>(
      ao, wpb, bproj, nullptr, nullptr, nullptr, out, M, DIMC, DIMC);
}
